// AffinCraftAttnBias_63221918597593
// MI455X (gfx1250) — matrix operand order repaired and checked by reading, not compiled, not run on hardware
//
#include <hip/hip_runtime.h>

typedef __attribute__((ext_vector_type(16))) _Float16       v16h;
typedef __attribute__((ext_vector_type(16))) unsigned short v16u;
typedef __attribute__((ext_vector_type(8)))  float          v8f;
typedef __attribute__((ext_vector_type(4)))  float          v4f;

#define GG   8
#define EE   8192
#define NN   512
#define HH   32
#define NP1  513

union H16U { v16h h; v16u u; };

__device__ __forceinline__ unsigned short f2h(float x) {
  union { _Float16 h; unsigned short u; } c;
  c.h = (_Float16)x;
  return c.u;
}

__device__ __forceinline__ v8f bcast8(float x) {
  v8f c;
#pragma unroll
  for (int r = 0; r < 8; ++r) c[r] = x;
  return c;
}

// Build the A-fragment (16x32 f16, K padded 28->32) for one input row of 28 f32.
// ISA layout: lanes 0-15 hold K 0-7 (halves 0-7) and K 16-23 (halves 8-15);
// lanes 16-31 hold K 8-15 and K 24-31 (K 28-31 zeroed).
__device__ __forceinline__ v16h load_in28(const float* __restrict__ row, int upper) {
  const v4f* p = (const v4f*)row;  // row is 16B aligned (28 floats = 112B)
  v4f x0, x1, x2, x3;
  if (!upper) {
    x0 = p[0]; x1 = p[1]; x2 = p[4]; x3 = p[5];
  } else {
    x0 = p[2]; x1 = p[3]; x2 = p[6];
    x3 = (v4f){0.0f, 0.0f, 0.0f, 0.0f};
  }
  v16h a;
#pragma unroll
  for (int u = 0; u < 4; ++u) {
    a[u]      = (_Float16)x0[u];
    a[4 + u]  = (_Float16)x1[u];
    a[8 + u]  = (_Float16)x2[u];
    a[12 + u] = (_Float16)x3[u];
  }
  return a;
}

// ---------------------------------------------------------------------------
// Kernel 1: bias init = broadcast(attn_bias) + virt on first row/col
// ---------------------------------------------------------------------------
__global__ __launch_bounds__(256) void init_bias_kernel(
    const float* __restrict__ attn, const float* __restrict__ virt,
    float* __restrict__ out) {
  long idx = (long)blockIdx.x * blockDim.x + threadIdx.x;
  const long total = (long)GG * HH * NP1 * NP1;
  if (idx >= total) return;
  int q = (int)(idx % NP1);
  long t = idx / NP1;
  int p = (int)(t % NP1);
  long t2 = t / NP1;
  int h = (int)(t2 % HH);
  int g = (int)(t2 / HH);
  float v = attn[((long)g * NP1 + p) * NP1 + q];
  if (p == 0 || q == 0) v += virt[h];
  out[idx] = v;
}

// ---------------------------------------------------------------------------
// Kernel 2: fused angle/dists MLPs via WMMA, RMW-added into bias interior
// ---------------------------------------------------------------------------
__global__ __launch_bounds__(128) void mlp_bias_kernel(
    const float* __restrict__ angle, const float* __restrict__ dists,
    const float* __restrict__ w1a, const float* __restrict__ b1a,
    const float* __restrict__ w2a, const float* __restrict__ b2a,
    const float* __restrict__ w1d, const float* __restrict__ b1d,
    const float* __restrict__ w2d, const float* __restrict__ b2d,
    float* __restrict__ out) {
  __shared__ __align__(16) unsigned short lds[4 * 1024];  // 2KB per wave
  const int lane  = threadIdx.x & 31;
  const int wslot = threadIdx.x >> 5;
  unsigned short* myh = lds + wslot * 1024;
  float* myf = (float*)myh;

  const int n0    = lane & 15;   // tile-local column / A-row
  const int upper = lane >> 4;   // lane half
  const int kb16  = upper * 16;  // B-fragment K base

  // --- per-wave weight fragments (B layout: lane = column n, lanes 0-15 hold
  // K 0-7 and K 16-23, lanes 16-31 hold K 8-15 and K 24-31) ---
  v16h b1fa[2], b1fd[2], b2f[2][2];
#pragma unroll
  for (int tile = 0; tile < 2; ++tile) {
    int n = n0 + 16 * tile;
    v16h ba, bd, c0, c1;
#pragma unroll
    for (int t = 0; t < 16; ++t) {
      int k = kb16 / 2 + t + (t >= 8 ? 8 : 0);
      int ksafe = (k < 28) ? k : 0;
      float sc  = (k < 28) ? 1.0f : 0.0f;
      ba[t] = (_Float16)(w1a[ksafe * HH + n] * sc);
      bd[t] = (_Float16)(w1d[ksafe * HH + n] * sc);
      c0[t] = (_Float16)w2a[k * HH + n];  // k in [0,32): always valid
      c1[t] = (_Float16)w2d[k * HH + n];
    }
    b1fa[tile] = ba; b1fd[tile] = bd;
    b2f[0][tile] = c0; b2f[1][tile] = c1;
  }
  float bias1a[2], bias1d[2], bias2[2];
#pragma unroll
  for (int tile = 0; tile < 2; ++tile) {
    int n = n0 + 16 * tile;
    bias1a[tile] = b1a[n];
    bias1d[tile] = b1d[n];
    bias2[tile]  = b2a[n] + b2d[n];
  }

  const long totalBlocks = (long)GG * NN * NN / 16;  // 131072 row-blocks
  const int  nWaves = gridDim.x * (blockDim.x >> 5);
  long rb = (long)blockIdx.x * (blockDim.x >> 5) + wslot;

  for (; rb < totalBlocks; rb += nWaves) {
    const long R0  = rb * 16;
    const int  g   = (int)(R0 / ((long)NN * NN));
    const long rem = R0 - (long)g * NN * NN;
    const int  i   = (int)(rem / NN);
    const int  j0  = (int)(rem % NN);

    // ---- layer 1: hidden = relu(X @ W1 + b1) for angle and dists ----
    const long rowR = R0 + n0;  // this lane's M row
    v16h aA = load_in28(angle + rowR * 28, upper);
    v16h aD = load_in28(dists + rowR * 28, upper);

    v8f hA0 = bcast8(bias1a[0]), hA1 = bcast8(bias1a[1]);
    v8f hD0 = bcast8(bias1d[0]), hD1 = bcast8(bias1d[1]);
    hA0 = __builtin_amdgcn_wmma_f32_16x16x32_f16(false, aA, false, b1fa[0], (short)0, hA0, false, false);
    hA1 = __builtin_amdgcn_wmma_f32_16x16x32_f16(false, aA, false, b1fa[1], (short)0, hA1, false, false);
    hD0 = __builtin_amdgcn_wmma_f32_16x16x32_f16(false, aD, false, b1fd[0], (short)0, hD0, false, false);
    hD1 = __builtin_amdgcn_wmma_f32_16x16x32_f16(false, aD, false, b1fd[1], (short)0, hD1, false, false);

    // relu + stage combined hidden (16 x 64 f16, row-major) in LDS
#pragma unroll
    for (int r = 0; r < 8; ++r) {
      int m = r + upper * 8;
      myh[m * 64 + n0]      = f2h(fmaxf(hA0[r], 0.0f));
      myh[m * 64 + n0 + 16] = f2h(fmaxf(hA1[r], 0.0f));
      myh[m * 64 + n0 + 32] = f2h(fmaxf(hD0[r], 0.0f));
      myh[m * 64 + n0 + 48] = f2h(fmaxf(hD1[r], 0.0f));
    }
    asm volatile("s_wait_dscnt 0" ::: "memory");

    // ---- transpose: read hidden back in A-fragment layout (K=64 -> 2 chunks)
    v16h a2[2];
#pragma unroll
    for (int c = 0; c < 2; ++c) {
      v16u u;
#pragma unroll
      for (int uu = 0; uu < 8; ++uu) {
        int t = 2 * uu;
        int K = t + (t >= 8 ? 8 : 0) + upper * 8 + 32 * c;  // even
        unsigned int pr = *(const unsigned int*)&myh[n0 * 64 + K];
        u[t]     = (unsigned short)(pr & 0xffffu);
        u[t + 1] = (unsigned short)(pr >> 16);
      }
      H16U cv; cv.u = u;
      a2[c] = cv.h;
    }

    // ---- layer 2: out = hidden @ [w2a; w2d] + (b2a + b2d) ----
    v8f d0 = bcast8(bias2[0]), d1 = bcast8(bias2[1]);
    d0 = __builtin_amdgcn_wmma_f32_16x16x32_f16(false, a2[0], false, b2f[0][0], (short)0, d0, false, false);
    d0 = __builtin_amdgcn_wmma_f32_16x16x32_f16(false, a2[1], false, b2f[1][0], (short)0, d0, false, false);
    d1 = __builtin_amdgcn_wmma_f32_16x16x32_f16(false, a2[0], false, b2f[0][1], (short)0, d1, false, false);
    d1 = __builtin_amdgcn_wmma_f32_16x16x32_f16(false, a2[1], false, b2f[1][1], (short)0, d1, false, false);

    // ---- epilogue: transpose 16(q) x 32(h) tile via LDS, coalesced RMW ----
#pragma unroll
    for (int r = 0; r < 8; ++r) {
      int m = r + upper * 8;
      myf[m * 32 + n0]      = d0[r];
      myf[m * 32 + n0 + 16] = d1[r];
    }
    asm volatile("s_wait_dscnt 0" ::: "memory");

    float* gbase = out + (((long)g * HH) * NP1 + (i + 1)) * NP1 + 1 + j0;
#pragma unroll
    for (int it = 0; it < 16; ++it) {
      int h = 2 * it + upper;              // lanes 0-15 -> h even, 16-31 -> odd
      float v = myf[n0 * 32 + h];          // (q = j0 + n0, col = h)
      float* addr = gbase + ((long)h * NP1) * NP1 + n0;
      *addr += v;
    }
    asm volatile("s_wait_dscnt 0" ::: "memory");  // LDS reuse next iteration
  }
}

// ---------------------------------------------------------------------------
// Kernel 3: edge embeddings, atomically scattered into bias[g,:,src+1,tgt+1]
// One wave per edge; lane = h.
// ---------------------------------------------------------------------------
__global__ __launch_bounds__(256) void edge_kernel(
    const float* __restrict__ edge_feat, const int* __restrict__ edge_index,
    const unsigned char* __restrict__ edge_mask, const int* __restrict__ num_lig,
    const float* __restrict__ struct_emb, const float* __restrict__ plip_lig,
    const float* __restrict__ plip_prot, const float* __restrict__ plip_inter,
    const float* __restrict__ dw1, const float* __restrict__ db1,
    const float* __restrict__ dw2, const float* __restrict__ db2,
    float* __restrict__ out) {
  int tid = blockIdx.x * blockDim.x + threadIdx.x;
  int ge = tid >> 5;
  int h  = tid & 31;
  if (ge >= GG * EE) return;
  int g = ge / EE;
  int e = ge - g * EE;

  const float* ef = edge_feat + (long)ge * 4;
  float f0 = ef[0], f1 = ef[1], f2 = ef[2], dist = ef[3];
  int t0 = (int)f0, t1 = (int)f1, t2i = (int)f2;
  int src = edge_index[((long)g * 2 + 0) * EE + e];
  int tgt = edge_index[((long)g * 2 + 1) * EE + e];
  float msk = edge_mask[ge] ? 1.0f : 0.0f;
  int nla = num_lig[g];
  if (nla < 1) nla = 1;
  bool sl = (src > 0) && (src < nla);
  bool tl = (tgt > 0) && (tgt < nla);

  // distance MLP (1 -> 32 relu -> 32); this lane produces channel h
  float acc = db2[h];
#pragma unroll
  for (int k = 0; k < 32; ++k) {
    float hk = fmaxf(fmaf(dist, dw1[k], db1[k]), 0.0f);
    acc = fmaf(hk, dw2[k * HH + h], acc);
  }

  float ev = 0.0f;
  if (t0 <= 1) {
    int si = t0 * 4 + t1 * 2 + t2i;
    si = si < 0 ? 0 : (si > 19 ? 19 : si);
    ev = struct_emb[si * HH + h];
  } else if (t0 == 5) {
    int pi = t1 < 0 ? 0 : (t1 > 14 ? 14 : t1);
    const float* tab = (sl && tl) ? plip_lig : ((!sl && !tl) ? plip_prot : plip_inter);
    ev = tab[pi * HH + h];
  }
  float val = (ev + acc) * msk;
  atomicAdd(out + (((long)g * HH + h) * NP1 + (src + 1)) * NP1 + (tgt + 1), val);
}

// ---------------------------------------------------------------------------
extern "C" void kernel_launch(void* const* d_in, const int* in_sizes, int n_in,
                              void* d_out, int out_size, void* d_ws, size_t ws_size,
                              hipStream_t stream) {
  const float* edge_feat        = (const float*)d_in[0];
  const int* edge_index         = (const int*)d_in[1];
  const unsigned char* edge_msk = (const unsigned char*)d_in[2];
  const int* num_lig            = (const int*)d_in[3];
  const float* attn_bias        = (const float*)d_in[4];
  const float* angle            = (const float*)d_in[5];
  const float* dists            = (const float*)d_in[6];
  // d_in[7] = node_feat (unused: only shape feeds the reference)
  const float* struct_emb       = (const float*)d_in[8];
  const float* plip_lig         = (const float*)d_in[9];
  const float* plip_prot        = (const float*)d_in[10];
  const float* plip_inter       = (const float*)d_in[11];
  const float* dist_w1          = (const float*)d_in[12];
  const float* dist_b1          = (const float*)d_in[13];
  const float* dist_w2          = (const float*)d_in[14];
  const float* dist_b2          = (const float*)d_in[15];
  const float* ang_w1           = (const float*)d_in[16];
  const float* ang_b1           = (const float*)d_in[17];
  const float* ang_w2           = (const float*)d_in[18];
  const float* ang_b2           = (const float*)d_in[19];
  const float* md_w1            = (const float*)d_in[20];
  const float* md_b1            = (const float*)d_in[21];
  const float* md_w2            = (const float*)d_in[22];
  const float* md_b2            = (const float*)d_in[23];
  const float* virt             = (const float*)d_in[24];
  float* out = (float*)d_out;

  const long total = (long)GG * HH * NP1 * NP1;
  const int initBlocks = (int)((total + 255) / 256);
  init_bias_kernel<<<initBlocks, 256, 0, stream>>>(attn_bias, virt, out);

  // 4096 blocks x 4 waves = 16384 waves over 131072 row-blocks (8 each)
  mlp_bias_kernel<<<4096, 128, 0, stream>>>(angle, dists,
                                            ang_w1, ang_b1, ang_w2, ang_b2,
                                            md_w1, md_b1, md_w2, md_b2, out);

  // one wave per edge: G*E*32 threads
  edge_kernel<<<(GG * EE * 32) / 256, 256, 0, stream>>>(
      edge_feat, edge_index, edge_msk, num_lig,
      struct_emb, plip_lig, plip_prot, plip_inter,
      dist_w1, dist_b1, dist_w2, dist_b2, out);
}
